// CollapsedPBFAOptimized_67035849556244
// MI455X (gfx1250) — matrix operand order repaired and checked by reading, not compiled, not run on hardware
//
#include <hip/hip_runtime.h>
#include <hip/hip_bf16.h>
#include <stdint.h>

// ---------------------------------------------------------------------------
// Collapsed-PBFA for MI455X (gfx1250, wave32, WMMA).
// Only Chebyshev orders 1..5 have nonzero beta -> NP=5 collapsed features.
// bf16 operands are pre-converted once; GEMMs use double-buffered LDS tiles
// staged with gfx1250 async global->LDS copies when the toolchain has them.
// ---------------------------------------------------------------------------

typedef __attribute__((ext_vector_type(16))) __bf16 v16bf;
typedef __attribute__((ext_vector_type(8)))  __bf16 v8bf;
typedef __attribute__((ext_vector_type(4)))  __bf16 v4bf;
typedef __attribute__((ext_vector_type(8)))  float  v8f;
typedef __attribute__((ext_vector_type(4)))  int    v4i;

#define K_B      2
#define K_S      1024
#define K_D      1024
#define K_H      16
#define K_DH     64
#define K_NP     5
#define K_NCHUNK 8
#define K_CHUNK  128
#define K_BH     (K_B * K_H)   // 32
#define K_TOK    (K_B * K_S)   // 2048

// gfx1250 async global->LDS copy path (ASYNCcnt), with portable fallback
#if defined(__has_builtin)
#if __has_builtin(__builtin_amdgcn_global_load_async_to_lds_b128) && \
    __has_builtin(__builtin_amdgcn_s_wait_asynccnt)
#define USE_ASYNC_COPY 1
#endif
#endif
#ifndef USE_ASYNC_COPY
#define USE_ASYNC_COPY 0
#endif

#if USE_ASYNC_COPY
typedef __attribute__((address_space(1))) v4i* gptr_v4i;   // global (AS1)
typedef __attribute__((address_space(3))) v4i* lptr_v4i;   // LDS    (AS3)
#endif

__device__ __forceinline__ void copy16_g2l(const __bf16* g, __bf16* l) {
#if USE_ASYNC_COPY
    __builtin_amdgcn_global_load_async_to_lds_b128((gptr_v4i)g, (lptr_v4i)l, 0, 0);
#else
    *(v8bf*)l = *(const v8bf*)g;
#endif
}

__device__ __forceinline__ void wait_async_all() {
#if USE_ASYNC_COPY
    __builtin_amdgcn_s_wait_asynccnt(0);
#endif
}

// beta[1..5] of the reference (beta[0]=beta[6..10]=0 after normalization)
__constant__ float c_beta[K_NP] = {0.457366f, 0.262187f, 0.155946f, 0.086940f, 0.037562f};

// Chebyshev T1..T5 (T0 contributes with beta=0, skipped)
__device__ __forceinline__ void cheb5(float x, float t[K_NP]) {
    const float x2 = 2.0f * x;
    t[0] = x;
    t[1] = x2 * x - 1.0f;
    t[2] = x2 * t[1] - t[0];
    t[3] = x2 * t[2] - t[1];
    t[4] = x2 * t[3] - t[2];
}

__device__ __forceinline__ void cheb5_acc(float x, float a[K_NP]) {
    float t[K_NP];
    cheb5(x, t);
#pragma unroll
    for (int p = 0; p < K_NP; ++p) a[p] += t[p];
}

// fp32 -> bf16 streaming conversion (done once; operands are reused many
// times by the GEMMs, so the cvt must not sit in the tile-staging path)
__global__ __launch_bounds__(256)
void f32_to_bf16(const float* __restrict__ src, __bf16* __restrict__ dst, int n4)
{
    const int i = (int)(blockIdx.x * blockDim.x + threadIdx.x);
    if (i >= n4) return;
    const float4 v = ((const float4*)src)[i];
    v4bf o = {(__bf16)v.x, (__bf16)v.y, (__bf16)v.z, (__bf16)v.w};
    ((v4bf*)dst)[i] = o;
}

// ---------------------------------------------------------------------------
// bf16 WMMA GEMM:  C[M,N] = A[M,K] * B[N,K]^T   (both K-contiguous, "NT")
// Block = 256 threads (8 waves), tile 128x128, BK=32, double-buffered LDS.
// mode 0: qkv epilogue (scale+clip q,k; scatter to head-major xq/xk/vv)
// mode 1: plain fp32 store to outC
// ---------------------------------------------------------------------------
__global__ __launch_bounds__(256)
void gemm_bf16_wmma(const __bf16* __restrict__ A, const __bf16* __restrict__ Bw,
                    int M, int N, int K, int mode,
                    float* __restrict__ xq, float* __restrict__ xk,
                    float* __restrict__ vv, float* __restrict__ outC)
{
    // 40 bf16 row stride (80B): 16B-aligned rows, padded vs. bank conflicts
    __shared__ __align__(16) __bf16 ldsA[2][128 * 40];
    __shared__ __align__(16) __bf16 ldsB[2][128 * 40];

    const int t    = (int)threadIdx.x;
    const int lane = t & 31;
    const int w    = t >> 5;
    const int wm   = w & 1;   // 2 M-groups of 64 rows
    const int wn   = w >> 1;  // 4 N-groups of 32 cols
    const int mBase = (int)blockIdx.y * 128;
    const int nBase = (int)blockIdx.x * 128;

    const v8f vzero = {0.f, 0.f, 0.f, 0.f, 0.f, 0.f, 0.f, 0.f};
    v8f acc[4][2];
#pragma unroll
    for (int mi = 0; mi < 4; ++mi)
#pragma unroll
        for (int ni = 0; ni < 2; ++ni) acc[mi][ni] = vzero;

    // issue one 128x32 bf16 tile (A and B) into LDS buffer `buf`:
    // 512 16B chunks per matrix / 256 threads = 2 chunks each
    auto issue_tile = [&](int k0, int buf) {
#pragma unroll
        for (int i = 0; i < 2; ++i) {
            const int q   = t + i * 256;   // 0..511
            const int row = q >> 2;        // 0..127
            const int kq  = (q & 3) << 3;  // 0,8,16,24 bf16
            copy16_g2l(A  + (size_t)(mBase + row) * K + k0 + kq,
                       &ldsA[buf][row * 40 + kq]);
            copy16_g2l(Bw + (size_t)(nBase + row) * K + k0 + kq,
                       &ldsB[buf][row * 40 + kq]);
        }
    };

    const int nTiles = K / 32;
    issue_tile(0, 0);

    for (int it = 0; it < nTiles; ++it) {
        const int cur = it & 1;
        wait_async_all();      // my chunks of tile `it` have landed in LDS
        __syncthreads();       // everyone's chunks visible; buf[nxt] is free
        if (it + 1 < nTiles) issue_tile((it + 1) * 32, cur ^ 1);  // overlap

        // A fragment (16x32 bf16): lane l row = l&15; lanes<16 hold K
        // {0..7,16..23}, lanes>=16 hold K {8..15,24..31} -> 2x ds_load_b128.
        v16bf afrag[4];
#pragma unroll
        for (int mi = 0; mi < 4; ++mi) {
            const int r   = wm * 64 + mi * 16 + (lane & 15);
            const int klo = (lane & 16) ? 8 : 0;
            const v8bf lo = *(const v8bf*)&ldsA[cur][r * 40 + klo];
            const v8bf hi = *(const v8bf*)&ldsA[cur][r * 40 + klo + 16];
            v16bf a;
#pragma unroll
            for (int e = 0; e < 8; ++e) { a[e] = lo[e]; a[e + 8] = hi[e]; }
            afrag[mi] = a;
        }
        // B fragment (32x16 bf16): lane l col = l&15; same K order as A.
        v16bf bfrag[2];
#pragma unroll
        for (int ni = 0; ni < 2; ++ni) {
            const int c  = wn * 32 + ni * 16 + (lane & 15);
            const int kh = (lane & 16) ? 8 : 0;
            const v8bf lo = *(const v8bf*)&ldsB[cur][c * 40 + kh];
            const v8bf hi = *(const v8bf*)&ldsB[cur][c * 40 + kh + 16];
            v16bf b;
#pragma unroll
            for (int e = 0; e < 8; ++e) { b[e] = lo[e]; b[e + 8] = hi[e]; }
            bfrag[ni] = b;
        }

#pragma unroll
        for (int mi = 0; mi < 4; ++mi)
#pragma unroll
            for (int ni = 0; ni < 2; ++ni)
                acc[mi][ni] = __builtin_amdgcn_wmma_f32_16x16x32_bf16(
                    false, afrag[mi], false, bfrag[ni],
                    (short)0, acc[mi][ni], false, false);
    }

    // Epilogue: C/D layout -> VGPR r, lanes 0-15: M=r, N=lane; lanes 16-31: M=r+8.
    const int mTop = mBase + wm * 64;
    const int nTop = nBase + wn * 32;
#pragma unroll
    for (int mi = 0; mi < 4; ++mi) {
#pragma unroll
        for (int ni = 0; ni < 2; ++ni) {
            const int n    = nTop + ni * 16 + (lane & 15);
            const int mOff = mTop + mi * 16 + ((lane & 16) ? 8 : 0);
#pragma unroll
            for (int r = 0; r < 8; ++r) {
                const int m    = mOff + r;
                const float vc = acc[mi][ni][r];
                if (mode == 0) {
                    // m = b*S+s ; n in [0,3072): 0..1023 q, 1024..2047 k, else v
                    const int b = m >> 10, s = m & 1023;
                    const int which = n >> 10;
                    const int h = (n >> 6) & 15;
                    const int d = n & 63;
                    const size_t o = ((size_t)(b * K_H + h) * K_S + s) * K_DH + d;
                    if (which == 2) {
                        vv[o] = vc;
                    } else {
                        float xc = vc * 0.125f;                 // 1/sqrt(64)
                        xc = fminf(fmaxf(xc, -0.999999f), 0.999999f);
                        if (which == 0) xq[o] = xc; else xk[o] = xc;
                    }
                } else {
                    outC[(size_t)m * N + n] = vc;
                }
            }
        }
    }
}

// ---------------------------------------------------------------------------
// Sq[bh,s,p] = sum_d Tq_p ; Sk[bh,s,p] = sum_d Tk_p   (one thread per (bh,s))
// ---------------------------------------------------------------------------
__global__ __launch_bounds__(256)
void feat_sums(const float* __restrict__ xq, const float* __restrict__ xk,
               float* __restrict__ Sq, float* __restrict__ Sk)
{
    const int idx = (int)(blockIdx.x * blockDim.x + threadIdx.x);   // bh*S + s
    if (idx >= K_BH * K_S) return;
    const float* qp = xq + (size_t)idx * K_DH;
    const float* kp = xk + (size_t)idx * K_DH;
    float sq[K_NP] = {0, 0, 0, 0, 0};
    float sk[K_NP] = {0, 0, 0, 0, 0};
#pragma unroll 4
    for (int d = 0; d < K_DH; d += 4) {
        const float4 q4 = *(const float4*)(qp + d);
        const float4 k4 = *(const float4*)(kp + d);
        cheb5_acc(q4.x, sq); cheb5_acc(q4.y, sq);
        cheb5_acc(q4.z, sq); cheb5_acc(q4.w, sq);
        cheb5_acc(k4.x, sk); cheb5_acc(k4.y, sk);
        cheb5_acc(k4.z, sk); cheb5_acc(k4.w, sk);
    }
#pragma unroll
    for (int p = 0; p < K_NP; ++p) {
        Sq[(size_t)idx * K_NP + p] = sq[p];
        Sk[(size_t)idx * K_NP + p] = sk[p];
    }
}

// ---------------------------------------------------------------------------
// ksb[bh,s,p] = beta_p * cumsum_s Sk  — one wave32 per (bh,p), shuffle scan
// ---------------------------------------------------------------------------
__global__ __launch_bounds__(160)
void ks_scan(const float* __restrict__ Sk, float* __restrict__ ksb)
{
    const int bh   = (int)blockIdx.x;       // 0..31
    const int p    = (int)threadIdx.x >> 5; // 0..4 (one wave per p)
    const int lane = (int)threadIdx.x & 31;
    const float beta = c_beta[p];
    float carry = 0.0f;
    for (int c = 0; c < K_S / 32; ++c) {
        const int s = c * 32 + lane;
        float v = Sk[((size_t)bh * K_S + s) * K_NP + p];
#pragma unroll
        for (int off = 1; off < 32; off <<= 1) {
            const float n = __shfl_up(v, (unsigned)off, 32);
            if (lane >= off) v += n;
        }
        const float total = __shfl(v, 31, 32);
        ksb[((size_t)bh * K_S + s) * K_NP + p] = (carry + v) * beta;
        carry += total;
    }
}

// den[bh,s] = sum_p Sq[bh,s,p] * ksb[bh,s,p]
__global__ __launch_bounds__(256)
void den_kernel(const float* __restrict__ Sq, const float* __restrict__ ksb,
                float* __restrict__ den)
{
    const int idx = (int)(blockIdx.x * blockDim.x + threadIdx.x);
    if (idx >= K_BH * K_S) return;
    float a = 0.0f;
#pragma unroll
    for (int p = 0; p < K_NP; ++p)
        a += Sq[(size_t)idx * K_NP + p] * ksb[(size_t)idx * K_NP + p];
    den[idx] = a;
}

// Per-chunk kv totals: ckv[bh,c,p,d] = sum_{s in chunk c} Tk_p(xk) * v
__global__ __launch_bounds__(256)
void chunk_kv(const float* __restrict__ xk, const float* __restrict__ vv,
              float* __restrict__ ckv)
{
    const int gid = (int)(blockIdx.x * blockDim.x + threadIdx.x);
    if (gid >= K_BH * K_NCHUNK * K_DH) return;
    const int d  = gid & 63;
    const int c  = (gid >> 6) & (K_NCHUNK - 1);
    const int bh = gid >> 9;
    float kv[K_NP] = {0, 0, 0, 0, 0};
    for (int s = c * K_CHUNK; s < (c + 1) * K_CHUNK; ++s) {
        const size_t o = ((size_t)bh * K_S + s) * K_DH + d;
        float t[K_NP];
        cheb5(xk[o], t);
        const float vval = vv[o];
#pragma unroll
        for (int p = 0; p < K_NP; ++p) kv[p] += t[p] * vval;
    }
#pragma unroll
    for (int p = 0; p < K_NP; ++p)
        ckv[(((size_t)bh * K_NCHUNK + c) * K_NP + p) * K_DH + d] = kv[p];
}

// Main scan: carry-in from prior chunks, then 128 sequential steps in
// registers; emits out_h directly in bf16 for the output-projection GEMM.
__global__ __launch_bounds__(256)
void attn_main(const float* __restrict__ xq, const float* __restrict__ xk,
               const float* __restrict__ vv, const float* __restrict__ ckv,
               const float* __restrict__ den, __bf16* __restrict__ outH)
{
    const int gid = (int)(blockIdx.x * blockDim.x + threadIdx.x);
    if (gid >= K_BH * K_NCHUNK * K_DH) return;
    const int d  = gid & 63;
    const int c  = (gid >> 6) & (K_NCHUNK - 1);
    const int bh = gid >> 9;
    const int b  = bh >> 4;
    const int h  = bh & 15;

    float kv[K_NP] = {0, 0, 0, 0, 0};
    for (int c2 = 0; c2 < c; ++c2)
#pragma unroll
        for (int p = 0; p < K_NP; ++p)
            kv[p] += ckv[(((size_t)bh * K_NCHUNK + c2) * K_NP + p) * K_DH + d];

    for (int s = c * K_CHUNK; s < (c + 1) * K_CHUNK; ++s) {
        const size_t o = ((size_t)bh * K_S + s) * K_DH + d;
        float tk[K_NP];
        cheb5(xk[o], tk);
        const float vval = vv[o];
#pragma unroll
        for (int p = 0; p < K_NP; ++p) kv[p] += tk[p] * vval;  // inclusive prefix

        float tq[K_NP];
        cheb5(xq[o], tq);
        float num = 0.0f;
#pragma unroll
        for (int p = 0; p < K_NP; ++p) num += tq[p] * kv[p] * c_beta[p];

        const float dn = den[(size_t)bh * K_S + s] + 1e-7f;
        outH[((size_t)b * K_S + s) * K_D + h * K_DH + d] = (__bf16)(num / dn);
    }
}

// ---------------------------------------------------------------------------
extern "C" void kernel_launch(void* const* d_in, const int* in_sizes, int n_in,
                              void* d_out, int out_size, void* d_ws, size_t ws_size,
                              hipStream_t stream)
{
    (void)in_sizes; (void)n_in; (void)out_size; (void)ws_size;
    const float* x     = (const float*)d_in[0];   // (2,1024,1024)
    const float* W_in  = (const float*)d_in[1];   // (3072,1024)
    const float* W_out = (const float*)d_in[2];   // (1024,1024)
    float* out = (float*)d_out;                   // (2,1024,1024)

    // workspace carve-up (all offsets 16B-aligned): ~42 MB total
    float* ws   = (float*)d_ws;
    float*  xq   = ws;                                          // 2,097,152 f32
    float*  xk   = xq   + (size_t)K_BH * K_S * K_DH;
    float*  vv   = xk   + (size_t)K_BH * K_S * K_DH;
    float*  Sq   = vv   + (size_t)K_BH * K_S * K_DH;            // 163,840 each
    float*  Sk   = Sq   + (size_t)K_BH * K_S * K_NP;
    float*  ksb  = Sk   + (size_t)K_BH * K_S * K_NP;
    float*  den  = ksb  + (size_t)K_BH * K_S * K_NP;            // 32,768
    float*  ckv  = den  + (size_t)K_BH * K_S;                   // 81,920
    __bf16* xb   = (__bf16*)(ckv + (size_t)K_BH * K_NCHUNK * K_NP * K_DH);
    __bf16* wb1  = xb  + (size_t)K_TOK * K_D;                   // bf16 x
    __bf16* wb2  = wb1 + (size_t)3 * K_D * K_D;                 // bf16 W_in
    __bf16* outH = wb2 + (size_t)K_D * K_D;                     // bf16 W_out
    // outH: K_TOK*K_D bf16

    // 0) one-time bf16 conversion of GEMM operands
    f32_to_bf16<<<(K_TOK * K_D / 4) / 256, 256, 0, stream>>>(x, xb, K_TOK * K_D / 4);
    f32_to_bf16<<<(3 * K_D * K_D / 4) / 256, 256, 0, stream>>>(W_in, wb1, 3 * K_D * K_D / 4);
    f32_to_bf16<<<(K_D * K_D / 4) / 256, 256, 0, stream>>>(W_out, wb2, K_D * K_D / 4);

    // 1) fused QKV projection + scale/clip + head transpose
    gemm_bf16_wmma<<<dim3(3 * K_D / 128, K_TOK / 128), 256, 0, stream>>>(
        xb, wb1, K_TOK, 3 * K_D, K_D, 0, xq, xk, vv, nullptr);

    // 2) per-position feature sums over Dh
    feat_sums<<<(K_BH * K_S) / 256, 256, 0, stream>>>(xq, xk, Sq, Sk);

    // 3) wave32 shuffle prefix-scan of Sk over S (x beta)
    ks_scan<<<K_BH, 32 * K_NP, 0, stream>>>(Sk, ksb);

    // 4) denominator
    den_kernel<<<(K_BH * K_S) / 256, 256, 0, stream>>>(Sq, ksb, den);

    // 5) chunked kv prefix: totals, then main scan with carry-in
    chunk_kv<<<(K_BH * K_NCHUNK * K_DH) / 256, 256, 0, stream>>>(xk, vv, ckv);
    attn_main<<<(K_BH * K_NCHUNK * K_DH) / 256, 256, 0, stream>>>(
        xq, xk, vv, ckv, den, outH);

    // 6) output projection
    gemm_bf16_wmma<<<dim3(K_D / 128, K_TOK / 128), 256, 0, stream>>>(
        outH, wb2, K_TOK, K_D, K_D, 1, nullptr, nullptr, nullptr, out);
}
